// AdaptiveAttention_69827578298919
// MI455X (gfx1250) — matrix operand order repaired and checked by reading, not compiled, not run on hardware
//
#include <hip/hip_runtime.h>

// Problem constants (match reference)
#define BB 4
#define TT 4096
#define DD 1024
#define SPAN_ 128
#define NBLK (TT / SPAN_) // 32
#define M_TOT (BB * TT)   // 16384

typedef __attribute__((ext_vector_type(16))) _Float16 v16h;
typedef __attribute__((ext_vector_type(8)))  _Float16 v8h;
typedef __attribute__((ext_vector_type(8)))  float    v8f;

union V16HU { v16h v; v8h h[2]; };

__device__ __forceinline__ v8f wmma_f16(v16h a, v16h b, v8f c) {
  // D = A(16x32 f16) * B(32x16 f16) + C(16x16 f32)
  return __builtin_amdgcn_wmma_f32_16x16x32_f16(false, a, false, b, (short)0, c,
                                                false, false);
}

__device__ __forceinline__ v8f vzero8() {
  v8f z;
#pragma unroll
  for (int i = 0; i < 8; ++i) z[i] = 0.f;
  return z;
}

// ---- gfx1250 async copy global->LDS (16B per lane), tracked by ASYNCcnt ----
__device__ __forceinline__ void async_ld16(unsigned ldsoff, const void* g) {
  asm volatile("global_load_async_to_lds_b128 %0, %1, off"
               :: "v"(ldsoff), "v"(g) : "memory");
}
__device__ __forceinline__ void wait_async0() {
  asm volatile("s_wait_asynccnt 0" ::: "memory");
}

// Read a 32B (16-half) fragment span from LDS as two 16B chunks.
__device__ __forceinline__ v16h lds_frag(const char* smem, unsigned byteoff) {
  V16HU u;
  u.h[0] = *reinterpret_cast<const v8h*>(smem + byteoff);
  u.h[1] = *reinterpret_cast<const v8h*>(smem + byteoff + 16);
  return u.v;
}

// A fragment (16x32 f16) from a row-major f16 row pointer (+k0 applied by caller)
// lane<16: halves {K0..15}; lane>=16: {K16..31}
__device__ __forceinline__ v16h a_frag_h(const _Float16* p, int hi) {
  V16HU u;
  u.h[0] = *reinterpret_cast<const v8h*>(p + hi * 16);
  u.h[1] = *reinterpret_cast<const v8h*>(p + hi * 16 + 8);
  return u.v;
}

// Swizzled LDS index for P[128][256] f16 (16B-chunk XOR swizzle, conflict-free)
__device__ __forceinline__ int pidx(int row, int col) {
  return (row << 8) + ((((col >> 3) ^ (row & 15)) << 3) | (col & 7));
}

// ---------------------------------------------------------------------------
// Kernel 0: fp32 -> f16 conversion of x and the three weight matrices.
// grid = (8192, 4); y selects source. 8 elements / thread.
// ---------------------------------------------------------------------------
__global__ void cvt_kernel(const float* __restrict__ x,
                           const float* __restrict__ Wq,
                           const float* __restrict__ Wk,
                           const float* __restrict__ Wv,
                           _Float16* __restrict__ xh,
                           _Float16* __restrict__ Whq,
                           _Float16* __restrict__ Whk,
                           _Float16* __restrict__ Whv) {
  const int y = blockIdx.y;
  const float* src = (y == 0) ? x : (y == 1) ? Wq : (y == 2) ? Wk : Wv;
  _Float16* dst    = (y == 0) ? xh : (y == 1) ? Whq : (y == 2) ? Whk : Whv;
  const size_t n = (y == 0) ? (size_t)M_TOT * DD : (size_t)DD * DD;
  const size_t i = ((size_t)blockIdx.x * blockDim.x + threadIdx.x) * 8;
  if (i >= n) return;
  const float4* p = reinterpret_cast<const float4*>(src + i);
  const float4 a = p[0], b = p[1];
  v8h o;
  o[0] = (_Float16)a.x; o[1] = (_Float16)a.y;
  o[2] = (_Float16)a.z; o[3] = (_Float16)a.w;
  o[4] = (_Float16)b.x; o[5] = (_Float16)b.y;
  o[6] = (_Float16)b.z; o[7] = (_Float16)b.w;
  *reinterpret_cast<v8h*>(dst + i) = o;
}

// ---------------------------------------------------------------------------
// Kernel 1: fused QKV projection, f16 WMMA. y = x @ W^T + b.
// grid = (DD/128, M_TOT/128, 3); block = 256 (8 waves), wave tile 32x64.
// B tile (128 W-rows x 32 k) double-buffered in LDS via async copies;
// A fragments software-pipelined in registers.
// Dynamic LDS: 2 * 10240 B (row stride 80 B = conflict-free).
// z==2 writes V transposed as VT[b][d][t].
// ---------------------------------------------------------------------------
__global__ void proj_kernel(const _Float16* __restrict__ xh,
                            const _Float16* __restrict__ Whq,
                            const _Float16* __restrict__ Whk,
                            const _Float16* __restrict__ Whv,
                            const float* __restrict__ bq,
                            const float* __restrict__ bk,
                            const float* __restrict__ bv,
                            _Float16* __restrict__ Qh,
                            _Float16* __restrict__ Kh,
                            _Float16* __restrict__ VT) {
  extern __shared__ char smem[]; // no static LDS: offsets start at 0
  const int z = blockIdx.z;
  const _Float16* W = (z == 0) ? Whq : (z == 1) ? Whk : Whv;
  const float* bias = (z == 0) ? bq : (z == 1) ? bk : bv;

  const int tid  = threadIdx.x;
  const int lane = tid & 31;
  const int w    = tid >> 5;   // 0..7
  const int wm   = w & 3;      // 4 waves along M
  const int wn   = w >> 2;     // 2 waves along N
  const int cl   = lane & 15;
  const int hi   = lane >> 4;

  const int m0  = blockIdx.y * 128 + wm * 32; // 2 M-tiles of 16
  const int nb0 = blockIdx.x * 128;           // 128 output cols in this WG
  const int n0  = nb0 + wn * 64;              // 4 N-tiles of 16 per wave

  // stage W[nb0..nb0+128)[k0..k0+32) -> LDS buffer (rows padded to 80B)
  auto stageB = [&](int bufsel, int k0) {
#pragma unroll
    for (int i = 0; i < 2; ++i) {
      const int c = tid + 256 * i;        // 512 chunks of 16B
      const int row = c >> 2, seg = c & 3;
      const _Float16* g = W + (size_t)(nb0 + row) * DD + k0 + seg * 8;
      async_ld16((unsigned)(bufsel * 10240 + row * 80 + seg * 16), g);
    }
  };

  const _Float16* arow0 = xh + (size_t)(m0 + cl) * DD;
  const _Float16* arow1 = xh + (size_t)(m0 + 16 + cl) * DD;

  v8f acc[2][4];
#pragma unroll
  for (int mi = 0; mi < 2; ++mi)
#pragma unroll
    for (int ni = 0; ni < 4; ++ni) acc[mi][ni] = vzero8();

  stageB(0, 0);
  v16h a0 = a_frag_h(arow0, hi);
  v16h a1 = a_frag_h(arow1, hi);

  int cur = 0;
  for (int k0 = 0; k0 < DD; k0 += 32) {
    wait_async0();
    __syncthreads();
    if (k0 + 32 < DD) stageB(cur ^ 1, k0 + 32);

    // prefetch next A fragments while this step computes
    const int kn = (k0 + 32 < DD) ? k0 + 32 : k0;
    const v16h a0n = a_frag_h(arow0 + kn, hi);
    const v16h a1n = a_frag_h(arow1 + kn, hi);

    // hoist all 4 B fragments, then run the 8-WMMA chain back-to-back
    v16h b[4];
#pragma unroll
    for (int ni = 0; ni < 4; ++ni)
      b[ni] = lds_frag(
          smem, (unsigned)(cur * 10240 + (wn * 64 + ni * 16 + cl) * 80 + hi * 32));

#pragma unroll
    for (int ni = 0; ni < 4; ++ni) {
      acc[0][ni] = wmma_f16(a0, b[ni], acc[0][ni]);
      acc[1][ni] = wmma_f16(a1, b[ni], acc[1][ni]);
    }

    a0 = a0n;
    a1 = a1n;
    cur ^= 1;
  }

  // Epilogue: bias add + f16 store (z==2: transposed scatter for V)
#pragma unroll
  for (int ni = 0; ni < 4; ++ni) {
    const int col = n0 + ni * 16 + cl;
    const float bval = bias[col];
#pragma unroll
    for (int mi = 0; mi < 2; ++mi) {
#pragma unroll
      for (int r = 0; r < 8; ++r) {
        const int mg = m0 + mi * 16 + hi * 8 + r;  // global row in [0, B*T)
        const float val = acc[mi][ni][r] + bval;
        if (z == 2) {
          const int bidx = mg >> 12;      // / TT
          const int t    = mg & (TT - 1); // % TT (128 | 4096: never crosses batch)
          VT[((size_t)(bidx * DD + col)) * TT + t] = (_Float16)val;
        } else {
          _Float16* o = (z == 0) ? Qh : Kh;
          o[(size_t)mg * DD + col] = (_Float16)val;
        }
      }
    }
  }
}

// ---------------------------------------------------------------------------
// Kernel 2: sliding-window attention over one (batch, query-block).
// grid = (NBLK, BB); block = 256 (8 waves). Wave w owns query rows
// [16w, 16w+16). Context = [prev block | cur block] (256 keys).
// Dynamic LDS: P (64KB, swizzled) + K-tile double buffer (2 x 20480 B).
// ---------------------------------------------------------------------------
#define KOFF 65536
#define KBUF 20480

__global__ void attn_kernel(const _Float16* __restrict__ Qh,
                            const _Float16* __restrict__ Kh,
                            const _Float16* __restrict__ VT,
                            float* __restrict__ out) {
  extern __shared__ char smem[];
  _Float16* P = reinterpret_cast<_Float16*>(smem);

  const int nblk = blockIdx.x;
  const int bb   = blockIdx.y;
  const int tid  = threadIdx.x;
  const int lane = tid & 31;
  const int w    = tid >> 5;
  const int cl   = lane & 15;
  const int hi   = lane >> 4;
  const int nctx0 = nblk * 128 - 128; // time index of context col 0

  // stage K context rows [0,256) x k-halves [k0,k0+32) into LDS
  auto stageK = [&](int bufsel, int k0) {
#pragma unroll
    for (int i = 0; i < 4; ++i) {
      const int c = tid + 256 * i;      // 1024 chunks of 16B
      const int row = c >> 2, seg = c & 3;
      int kt = nctx0 + row;
      if (kt < 0) kt = 0;               // n==0 padding rows: masked later
      const _Float16* g = Kh + ((size_t)(bb * TT + kt)) * DD + k0 + seg * 8;
      async_ld16((unsigned)(KOFF + bufsel * KBUF + row * 80 + seg * 16), g);
    }
  };

  // ---- Phase 1: S = Q K^T ----
  v8f accS[16];
#pragma unroll
  for (int t = 0; t < 16; ++t) accS[t] = vzero8();

  const _Float16* qrowp =
      Qh + ((size_t)(bb * TT + nblk * 128 + 16 * w + cl)) * DD;

  stageK(0, 0);
  v16h a = a_frag_h(qrowp, hi);

  int cur = 0;
  for (int k0 = 0; k0 < DD; k0 += 32) {
    wait_async0();
    __syncthreads();
    if (k0 + 32 < DD) stageK(cur ^ 1, k0 + 32);

    const int kn = (k0 + 32 < DD) ? k0 + 32 : k0;
    const v16h an = a_frag_h(qrowp + kn, hi); // prefetch next Q fragment

#pragma unroll
    for (int t = 0; t < 16; ++t) {
      const v16h b = lds_frag(
          smem, (unsigned)(KOFF + cur * KBUF + (16 * t + cl) * 80 + hi * 32));
      accS[t] = wmma_f16(a, b, accS[t]);
    }
    a = an;
    cur ^= 1;
  }

  // ---- masked softmax over 256 context cols, per query row ----
  const float scale = 0.03125f; // 1/sqrt(1024)
#pragma unroll
  for (int r = 0; r < 8; ++r) {
    const int qr = 16 * w + 8 * hi + r; // row within 128-block
    float vals[16];
    float mx = -1e30f;
#pragma unroll
    for (int t = 0; t < 16; ++t) {
      const int j = 16 * t + cl; // context col in [0,256)
      const bool valid =
          (j > qr) && (j <= qr + 128) && (nblk > 0 || j >= 128);
      const float s = valid ? accS[t][r] * scale : -1e30f;
      vals[t] = s;
      mx = fmaxf(mx, s);
    }
#pragma unroll
    for (int m = 8; m >= 1; m >>= 1) mx = fmaxf(mx, __shfl_xor(mx, m, 32));
    float sum = 0.f;
#pragma unroll
    for (int t = 0; t < 16; ++t) {
      const float e = __expf(vals[t] - mx);
      vals[t] = e;
      sum += e;
    }
#pragma unroll
    for (int m = 8; m >= 1; m >>= 1) sum += __shfl_xor(sum, m, 32);
    const float inv = 1.f / sum;
#pragma unroll
    for (int t = 0; t < 16; ++t)
      P[pidx(qr, 16 * t + cl)] = (_Float16)(vals[t] * inv);
  }
  __syncthreads();

  // ---- Phase 2: O = P V (8 chunks of 128 output cols) ----
#pragma unroll 1
  for (int nc = 0; nc < 8; ++nc) {
    v8f acc2[8];
#pragma unroll
    for (int nt = 0; nt < 8; ++nt) acc2[nt] = vzero8();

    for (int ks = 0; ks < 8; ++ks) {
      const int k0 = ks * 32;
      const int prow = 16 * w + cl;
      V16HU ua;
      ua.h[0] = *reinterpret_cast<const v8h*>(&P[pidx(prow, k0 + hi * 16)]);
      ua.h[1] = *reinterpret_cast<const v8h*>(&P[pidx(prow, k0 + hi * 16 + 8)]);
      const v16h ap = ua.v;

      int kt0 = nctx0 + k0 + hi * 16;   // key time base for this fragment
      if (kt0 < 0) kt0 = 0;             // n==0: P is 0 there anyway

      // hoist all 8 V fragments (contiguous VT rows), then 8 WMMAs
      v16h b[8];
#pragma unroll
      for (int nt = 0; nt < 8; ++nt) {
        const int col = nc * 128 + nt * 16 + cl;
        b[nt] = *reinterpret_cast<const v16h*>(
            VT + ((size_t)(bb * DD + col)) * TT + kt0);
      }
#pragma unroll
      for (int nt = 0; nt < 8; ++nt)
        acc2[nt] = wmma_f16(ap, b[nt], acc2[nt]);
    }

#pragma unroll
    for (int nt = 0; nt < 8; ++nt) {
      const int col = nc * 128 + nt * 16 + cl;
#pragma unroll
      for (int r = 0; r < 8; ++r) {
        const int trow = nblk * 128 + 16 * w + 8 * hi + r;
        out[((size_t)(bb * TT + trow)) * DD + col] = acc2[nt][r];
      }
    }
  }
}

// ---------------------------------------------------------------------------
extern "C" void kernel_launch(void* const* d_in, const int* in_sizes, int n_in,
                              void* d_out, int out_size, void* d_ws, size_t ws_size,
                              hipStream_t stream) {
  const float* x  = (const float*)d_in[0];
  const float* Wq = (const float*)d_in[1];
  const float* bq = (const float*)d_in[2];
  const float* Wk = (const float*)d_in[3];
  const float* bk = (const float*)d_in[4];
  const float* Wv = (const float*)d_in[5];
  const float* bv = (const float*)d_in[6];
  float* out = (float*)d_out;

  // Workspace (f16): Qh | Kh | VT | xh (each [B*T*D]) | Whq | Whk | Whv (D*D)
  _Float16* Qh  = (_Float16*)d_ws;
  _Float16* Kh  = Qh + (size_t)M_TOT * DD;
  _Float16* VT  = Kh + (size_t)M_TOT * DD;
  _Float16* xh  = VT + (size_t)M_TOT * DD;
  _Float16* Whq = xh + (size_t)M_TOT * DD;
  _Float16* Whk = Whq + (size_t)DD * DD;
  _Float16* Whv = Whk + (size_t)DD * DD;

  cvt_kernel<<<dim3((M_TOT * (size_t)DD) / (256 * 8), 4), 256, 0, stream>>>(
      x, Wq, Wk, Wv, xh, Whq, Whk, Whv);
  proj_kernel<<<dim3(DD / 128, M_TOT / 128, 3), 256, 2 * 10240, stream>>>(
      xh, Whq, Whk, Whv, bq, bk, bv, Qh, Kh, VT);
  attn_kernel<<<dim3(NBLK, BB), 256, KOFF + 2 * KBUF, stream>>>(Qh, Kh, VT, out);
}
